// UPNBase_31817117729082
// MI455X (gfx1250) — matrix operand order repaired and checked by reading, not compiled, not run on hardware
//
#include <hip/hip_runtime.h>
#include <hip/hip_bf16.h>

typedef __attribute__((ext_vector_type(16))) _Float16 v16h;
typedef __attribute__((ext_vector_type(8)))  _Float16 v8h;
typedef __attribute__((ext_vector_type(8)))  float    v8f;

#define BATCH 4096
#define DDIM  64
#define HDIM  512
#define BS_LD 520   // padded LDS row stride (halves): 1040 B = 260 dwords, 260%64=4

// ---------------- fragment loaders (wave32 WMMA 16x16x32 f16) ----------------
// A tile (16xK row-major, M = lane&15): 16 contiguous k-halves (k = 16h + s).
__device__ __forceinline__ v16h load_a_frag(const _Float16* __restrict__ Atile,
                                            int ldk, int k0, int lane) {
  const _Float16* p = Atile + (lane & 15) * ldk + k0 + ((lane >> 4) << 4);
  v8h lo = *(const v8h*)(p);
  v8h hi = *(const v8h*)(p + 8);
  return __builtin_shufflevector(lo, hi, 0,1,2,3,4,5,6,7,8,9,10,11,12,13,14,15);
}

// B tile: column n of B == row n of W (N x K row-major); 16 contiguous k-halves.
__device__ __forceinline__ v16h load_b_frag(const _Float16* __restrict__ Wtile,
                                            int ldk, int k0, int lane) {
  const _Float16* p = Wtile + (lane & 15) * ldk + k0 + ((lane >> 4) << 4);
  return *(const v16h*)p;
}

// ---------------- prep: fp32 -> f16 weight/activation staging ----------------
__global__ void prep_kernel(const float* __restrict__ states,
                            const float* __restrict__ W1, const float* __restrict__ W2,
                            const float* __restrict__ W3, const float* __restrict__ V1,
                            const float* __restrict__ V2, const float* __restrict__ V3,
                            _Float16* __restrict__ W1h, _Float16* __restrict__ W2h,
                            _Float16* __restrict__ W3h, _Float16* __restrict__ V1h,
                            _Float16* __restrict__ V2h, _Float16* __restrict__ V3h,
                            _Float16* __restrict__ W1th, _Float16* __restrict__ muh) {
  const int tid = blockIdx.x * blockDim.x + threadIdx.x;
  const int nt  = gridDim.x * blockDim.x;
  for (int i = tid; i < HDIM * DDIM; i += nt) {          // W1 (512x64), V1, W1^T
    float w = W1[i];
    W1h[i] = (_Float16)w;
    int k = i >> 6, d = i & 63;
    W1th[d * HDIM + k] = (_Float16)w;
    V1h[i] = (_Float16)V1[i];
  }
  for (int i = tid; i < HDIM * HDIM; i += nt) {          // W2, V2 (512x512)
    W2h[i] = (_Float16)W2[i];
    V2h[i] = (_Float16)V2[i];
  }
  for (int i = tid; i < DDIM * HDIM; i += nt) {          // W3, V3 (64x512)
    W3h[i] = (_Float16)W3[i];
    V3h[i] = (_Float16)V3[i];
  }
  for (int i = tid; i < BATCH * DDIM; i += nt) {         // mu = states[:, :64]
    int b = i >> 6, d = i & 63;
    muh[i] = (_Float16)states[b * 128 + d];
  }
}

// ---------------- generic WMMA GEMM + activation: C = act(A @ W^T + bias) ----
// A: MxK f16 row-major, W: NxK f16 row-major. One wave -> 16(M) x 64(N) strip.
// mode 0: tanh -> out_h f16; derivative 1-t^2 -> out_dh (f16) / out_f (f32, ld N)
// mode 1: linear -> out_f (ld = ldf)
// mode 2: softplus -> out_f (ld = ldf)
__global__ void gemm_act_kernel(const _Float16* __restrict__ A,
                                const _Float16* __restrict__ W,
                                const float* __restrict__ bias,
                                int M, int N, int K, int mode,
                                _Float16* __restrict__ out_h,
                                _Float16* __restrict__ out_dh,
                                float*    __restrict__ out_f,
                                int ldf) {
  const int lane  = threadIdx.x & 31;
  const int wave  = threadIdx.x >> 5;
  const int nblk  = N >> 6;
  const int strip = blockIdx.x * (blockDim.x >> 5) + wave;
  const int total = (M >> 4) * nblk;
  if (strip >= total) return;                 // wave-uniform: EXEC stays full
  const int m0 = (strip / nblk) << 4;
  const int n0 = (strip % nblk) << 6;
  const int nl = lane & 15;

  v8f acc[4];
#pragma unroll
  for (int j = 0; j < 4; ++j) {
    float bv = bias[n0 + 16 * j + nl];
    v8f t = {bv, bv, bv, bv, bv, bv, bv, bv};
    acc[j] = t;
  }

  const _Float16* Atile = A + m0 * K;
  for (int k0 = 0; k0 < K; k0 += 32) {
    v16h a = load_a_frag(Atile, K, k0, lane);
#pragma unroll
    for (int j = 0; j < 4; ++j) {
      v16h b = load_b_frag(W + (n0 + 16 * j) * K, K, k0, lane);
      acc[j] = __builtin_amdgcn_wmma_f32_16x16x32_f16(
          false, a, false, b, (short)0, acc[j], false, false);
    }
  }

#pragma unroll
  for (int j = 0; j < 4; ++j) {
    const int n = n0 + 16 * j + nl;
#pragma unroll
    for (int r = 0; r < 8; ++r) {
      const int m = m0 + r + ((lane >> 4) << 3);
      float x = acc[j][r];
      if (mode == 0) {
        float t = tanhf(x);
        out_h[m * N + n] = (_Float16)t;
        float dd = 1.f - t * t;
        if (out_dh) out_dh[m * N + n] = (_Float16)dd;
        if (out_f)  out_f[m * N + n]  = dd;
      } else if (mode == 1) {
        out_f[m * ldf + n] = x;
      } else {
        float sp = (x > 20.f) ? x : log1pf(__expf(x));
        out_f[m * ldf + n] = sp;
      }
    }
  }
}

// ---------------- Jacobian diagonal + sigma drift ----------------------------
// One workgroup per batch b (512 threads = 16 waves; wave w: h in [32w, 32w+32)).
// Y[b] = W2 @ (diag(d1[b]) * W1)   (512x64, K=512)  via WMMA, B staged in LDS
// with padded rows to avoid bank conflicts.
// J_diag[b,d] = sum_h d2[b,h] * W3[d,h] * Y[b,h,d]
// out[b,64+d] = 2*J*sigma + Q
__global__ void jdiag_kernel(const _Float16* __restrict__ W2h,
                             const _Float16* __restrict__ W1th,   // 64x512 (d-major)
                             const _Float16* __restrict__ d1h,    // Bx512
                             const float*    __restrict__ d2f,    // Bx512
                             const float*    __restrict__ W3,     // 64x512 fp32
                             const float*    __restrict__ Qd,     // Bx64
                             const float*    __restrict__ states, // Bx128
                             float*          __restrict__ out) {
  __shared__ __align__(32) _Float16 Bs[DDIM * BS_LD];  // d1-scaled W1^T (padded)
  __shared__ float jacc[DDIM];

  const int b = blockIdx.x;
  const int t = threadIdx.x;   // 512 threads
  if (t < DDIM) jacc[t] = 0.f;

  {  // stage B operand: Bs[d][k] = W1t[d][k] * d1[b][k]  (8 threads per row)
    const int d = t >> 3;
    const int kbase = (t & 7) * 64;
    const _Float16* wrow = W1th + d * HDIM;
    const _Float16* drow = d1h + b * HDIM;
    _Float16* srow = Bs + d * BS_LD;
#pragma unroll
    for (int k = kbase; k < kbase + 64; k += 8) {
      v8h wv = *(const v8h*)(wrow + k);
      v8h dv = *(const v8h*)(drow + k);
      v8h ov = wv * dv;
      *(v8h*)(srow + k) = ov;
    }
  }
  __syncthreads();

  const int lane = t & 31;
  const int wave = t >> 5;   // 0..15
  const int h0 = wave * 32;  // this wave: h in [h0, h0+32), d in [0,64)

  v8f acc[2][4];
  {
    v8f z = {0.f, 0.f, 0.f, 0.f, 0.f, 0.f, 0.f, 0.f};
#pragma unroll
    for (int rt = 0; rt < 2; ++rt)
#pragma unroll
      for (int jd = 0; jd < 4; ++jd) acc[rt][jd] = z;
  }

  for (int k0 = 0; k0 < HDIM; k0 += 32) {
    v16h bfr[4];
#pragma unroll
    for (int jd = 0; jd < 4; ++jd) {
      const _Float16* p =
          Bs + (jd * 16 + (lane & 15)) * BS_LD + k0 + ((lane >> 4) << 4);
      v8h lo = *(const v8h*)(p);       // ds_load_b128
      v8h hi = *(const v8h*)(p + 8);   // ds_load_b128
      bfr[jd] = __builtin_shufflevector(lo, hi,
                                        0,1,2,3,4,5,6,7,8,9,10,11,12,13,14,15);
    }
#pragma unroll
    for (int rt = 0; rt < 2; ++rt) {
      v16h a = load_a_frag(W2h + (h0 + rt * 16) * HDIM, HDIM, k0, lane);
#pragma unroll
      for (int jd = 0; jd < 4; ++jd)
        acc[rt][jd] = __builtin_amdgcn_wmma_f32_16x16x32_f16(
            false, a, false, bfr[jd], (short)0, acc[rt][jd], false, false);
    }
  }

  // reduce over h with the d2*W3 scale
  float part[4] = {0.f, 0.f, 0.f, 0.f};
#pragma unroll
  for (int rt = 0; rt < 2; ++rt) {
#pragma unroll
    for (int r = 0; r < 8; ++r) {
      const int h = h0 + rt * 16 + r + ((lane >> 4) << 3);
      const float s2 = d2f[b * HDIM + h];
#pragma unroll
      for (int jd = 0; jd < 4; ++jd) {
        const int d = jd * 16 + (lane & 15);
        part[jd] += acc[rt][jd][r] * s2 * W3[d * HDIM + h];
      }
    }
  }
#pragma unroll
  for (int jd = 0; jd < 4; ++jd) {
    const int d = jd * 16 + (lane & 15);
    atomicAdd(&jacc[d], part[jd]);              // ds_add_f32
  }
  __syncthreads();

  if (t < DDIM) {
    const float J  = jacc[t];
    const float sd = states[b * 128 + 64 + t];
    out[b * 128 + 64 + t] = 2.f * J * sd + Qd[b * DDIM + t];
  }
}

// ---------------- launcher ---------------------------------------------------
extern "C" void kernel_launch(void* const* d_in, const int* in_sizes, int n_in,
                              void* d_out, int out_size, void* d_ws, size_t ws_size,
                              hipStream_t stream) {
  const float* states = (const float*)d_in[1];
  const float* W1 = (const float*)d_in[2];
  const float* b1 = (const float*)d_in[3];
  const float* W2 = (const float*)d_in[4];
  const float* b2 = (const float*)d_in[5];
  const float* W3 = (const float*)d_in[6];
  const float* b3 = (const float*)d_in[7];
  const float* V1 = (const float*)d_in[8];
  const float* c1 = (const float*)d_in[9];
  const float* V2 = (const float*)d_in[10];
  const float* c2 = (const float*)d_in[11];
  const float* V3 = (const float*)d_in[12];
  const float* c3 = (const float*)d_in[13];
  float* out = (float*)d_out;

  char* ws = (char*)d_ws;
  size_t off = 0;
  auto alloc = [&](size_t bytes) -> char* {
    char* p = ws + off;
    off += (bytes + 255) & ~(size_t)255;
    return p;
  };
  _Float16* W1h  = (_Float16*)alloc((size_t)HDIM * DDIM * 2);
  _Float16* W2h  = (_Float16*)alloc((size_t)HDIM * HDIM * 2);
  _Float16* W3h  = (_Float16*)alloc((size_t)DDIM * HDIM * 2);
  _Float16* V1h  = (_Float16*)alloc((size_t)HDIM * DDIM * 2);
  _Float16* V2h  = (_Float16*)alloc((size_t)HDIM * HDIM * 2);
  _Float16* V3h  = (_Float16*)alloc((size_t)DDIM * HDIM * 2);
  _Float16* W1th = (_Float16*)alloc((size_t)DDIM * HDIM * 2);
  _Float16* muh  = (_Float16*)alloc((size_t)BATCH * DDIM * 2);
  _Float16* h1h  = (_Float16*)alloc((size_t)BATCH * HDIM * 2);
  _Float16* d1h  = (_Float16*)alloc((size_t)BATCH * HDIM * 2);
  _Float16* h2h  = (_Float16*)alloc((size_t)BATCH * HDIM * 2);
  float*    d2f  = (float*)   alloc((size_t)BATCH * HDIM * 4);
  _Float16* g1h  = (_Float16*)alloc((size_t)BATCH * HDIM * 2);
  _Float16* g2h  = (_Float16*)alloc((size_t)BATCH * HDIM * 2);
  float*    Qd   = (float*)   alloc((size_t)BATCH * DDIM * 4);

  prep_kernel<<<dim3(256), dim3(256), 0, stream>>>(
      states, W1, W2, W3, V1, V2, V3, W1h, W2h, W3h, V1h, V2h, V3h, W1th, muh);

  // strips = (M/16)*(N/64); 8 waves per 256-thread block
  // h1 = tanh(mu @ W1^T + b1); keep tanh (f16) and 1-t^2 (f16)
  gemm_act_kernel<<<dim3(256), dim3(256), 0, stream>>>(
      muh, W1h, b1, BATCH, HDIM, DDIM, 0, h1h, d1h, nullptr, 0);
  // h2 = tanh(h1 @ W2^T + b2); keep tanh (f16) and 1-t^2 (f32)
  gemm_act_kernel<<<dim3(256), dim3(256), 0, stream>>>(
      h1h, W2h, b2, BATCH, HDIM, HDIM, 0, h2h, nullptr, d2f, 0);
  // g1, g2
  gemm_act_kernel<<<dim3(256), dim3(256), 0, stream>>>(
      muh, V1h, c1, BATCH, HDIM, DDIM, 0, g1h, nullptr, nullptr, 0);
  gemm_act_kernel<<<dim3(256), dim3(256), 0, stream>>>(
      g1h, V2h, c2, BATCH, HDIM, HDIM, 0, g2h, nullptr, nullptr, 0);
  // mu_drift -> out[:, 0:64] (row stride 128)
  gemm_act_kernel<<<dim3(32), dim3(256), 0, stream>>>(
      h2h, W3h, b3, BATCH, DDIM, HDIM, 1, nullptr, nullptr, out, 128);
  // Q_diag = softplus(g2 @ V3^T + c3)
  gemm_act_kernel<<<dim3(32), dim3(256), 0, stream>>>(
      g2h, V3h, c3, BATCH, DDIM, HDIM, 2, nullptr, nullptr, Qd, DDIM);
  // Jacobian diagonal + sigma_drift -> out[:, 64:128]
  jdiag_kernel<<<dim3(BATCH), dim3(512), 0, stream>>>(
      W2h, W1th, d1h, d2f, W3, Qd, states, out);

  (void)in_sizes; (void)n_in; (void)out_size; (void)ws_size;
}
